// X_Feat_3058016714954
// MI455X (gfx1250) — matrix operand order repaired and checked by reading, not compiled, not run on hardware
//
#include <hip/hip_runtime.h>
#include <hip/hip_bf16.h>

typedef __attribute__((ext_vector_type(16))) _Float16 v16h;
typedef __attribute__((ext_vector_type(8)))  _Float16 h8;
typedef __attribute__((ext_vector_type(8)))  float    v8f;

#define NSEQ 4096
#define CDIM 128
#define BATCH 8
// feats columns: 0..1023 = f1 (b*128+c), 1024..1151 = f2 (1024 + b*16 + c)
#define FCOLS 1152
// per-batch padded feature depth for the epilogue GEMM (144 -> 160)
#define KPAD 160

__device__ __forceinline__ v8f wmma_f16(v16h a, v16h b, v8f c) {
  // D = A(16x32 f16) * B(32x16 f16) + C(16x16 f32)
  return __builtin_amdgcn_wmma_f32_16x16x32_f16(
      /*neg_a=*/false, a, /*neg_b=*/false, b,
      /*c_mod=*/(short)0, c, /*reuse_a=*/false, /*reuse_b=*/false);
}

// ---------------------------------------------------------------------------
// Kernel 1: Ht[col][j] (f16), col-major H panel so WMMA B loads are contiguous.
//   col in [0,1024): b*128+c  -> h1[b][j][c] = x[b][j]@W1[:,c] + b1[c]
//   col in [1024,1152): 1024+b*16+c2 -> h2[b][j][c2] = x[b][j]@W2[:,c2] + b2[c2]
// one block per (b, j); 160 threads (5 waves)
// ---------------------------------------------------------------------------
__global__ void prep_ht(const float* __restrict__ x,
                        const float* __restrict__ W1, const float* __restrict__ b1,
                        const float* __restrict__ W2, const float* __restrict__ b2,
                        _Float16* __restrict__ Ht) {
  __shared__ float xs[CDIM];
  const int bj = blockIdx.x;          // b*4096 + j
  const int b  = bj >> 12;
  const int j  = bj & (NSEQ - 1);
  const int t  = threadIdx.x;
  if (t < CDIM) xs[t] = x[(size_t)bj * CDIM + t];
  __syncthreads();
  if (t < 144) {
    float sum;
    int col;
    if (t < CDIM) {
      sum = b1[t];
      #pragma unroll 8
      for (int kk = 0; kk < CDIM; ++kk) sum += xs[kk] * W1[kk * CDIM + t];
      col = b * 128 + t;
    } else {
      const int c2 = t - CDIM;
      sum = b2[c2];
      #pragma unroll 8
      for (int kk = 0; kk < CDIM; ++kk) sum += xs[kk] * W2[kk * 16 + c2];
      col = 1024 + b * 16 + c2;
    }
    Ht[(size_t)col * NSEQ + j] = (_Float16)sum;
  }
}

// ---------------------------------------------------------------------------
// Kernel 2: Wbt[n][kk] f16, kk in [0,160): rows of Wb (144) zero-padded to 160.
// ---------------------------------------------------------------------------
__global__ void prep_wbt(const float* __restrict__ Wb, _Float16* __restrict__ Wbt) {
  const int idx = blockIdx.x * blockDim.x + threadIdx.x;
  if (idx >= 128 * KPAD) return;
  const int n = idx / KPAD, kk = idx % KPAD;
  const float v = (kk < 144) ? Wb[kk * 128 + n] : 0.0f;
  Wbt[idx] = (_Float16)v;
}

// ---------------------------------------------------------------------------
// Kernel 3: fused mask-GEMM + output GEMM.
// Grid: 128 blocks (32 rows each) x 256 threads (8 waves; wave w = batch b).
// Stage 1: feats[i, :] = [m1 @ H1 | m2 @ H2] for a 32-row tile, batches folded
//          into columns; masks rebuilt in-register from one sp tile per wave.
// Stage 2 (per 16-row half): feats tile -> LDS (f16, per-batch K=160 padded),
//          out[b,i,:] = featsA @ Wbt + bb via a second WMMA chain.
// ---------------------------------------------------------------------------
__global__ void __launch_bounds__(256)
masked_gemm_fused(const int* __restrict__ sp,
                  const _Float16* __restrict__ Ht,
                  const _Float16* __restrict__ Wbt,
                  const float* __restrict__ bb,
                  float* __restrict__ out) {
  __shared__ __attribute__((aligned(32))) _Float16 fl[16 * BATCH * KPAD]; // 40 KB

  const int tid   = threadIdx.x;
  const int b     = tid >> 5;          // wave index == batch
  const int lane  = tid & 31;
  const int half  = lane >> 4;         // lane group (0: lanes 0-15, 1: 16-31)
  const int lr    = lane & 15;
  const int ibase = blockIdx.x * 32;

  const int koffA = half ? 16 : 0;     // A-matrix K offset per lane group
  const int koffB = half ? 16 : 0;     // B-matrix K offset per lane group

  v8f acc1[2][8];                      // f1 tiles: 2 row-tiles x 8 col-tiles
  v8f acc2[2];                         // f2 tile per row-tile
  #pragma unroll
  for (int rt = 0; rt < 2; ++rt) {
    #pragma unroll
    for (int t = 0; t < 8; ++t) acc1[rt][t] = (v8f){};
    acc2[rt] = (v8f){};
  }

  const int row0 = ibase + lr;         // A row for row-tile 0 (this lane)
  const int row1 = ibase + 16 + lr;    // A row for row-tile 1

  for (int k = 0; k < NSEQ; k += 32) {
    // ---- build mask A fragments from sp (row per lane, contiguous K) ----
    v16h a1[2], a2[2];
    #pragma unroll
    for (int rt = 0; rt < 2; ++rt) {
      const int row = rt ? row1 : row0;
      const int* pr = sp + (size_t)row * NSEQ + k + koffA;
      const int4 pA = *(const int4*)(pr);
      const int4 pB = *(const int4*)(pr + 4);
      const int4 pC = *(const int4*)(pr + 8);
      const int4 pD = *(const int4*)(pr + 12);
      int sv[16] = {pA.x, pA.y, pA.z, pA.w, pB.x, pB.y, pB.z, pB.w,
                    pC.x, pC.y, pC.z, pC.w, pD.x, pD.y, pD.z, pD.w};
      #pragma unroll
      for (int e = 0; e < 16; ++e) {
        a1[rt][e] = (_Float16)(sv[e] <= 1 ? 1.0f : 0.0f);  // m1: sp <= 1
        a2[rt][e] = (_Float16)(sv[e] == 2 ? 1.0f : 0.0f);  // m2: sp == 2
      }
    }
    // ---- f1 column tiles (batch b, 128 cols): B shared across row-tiles ----
    #pragma unroll
    for (int t = 0; t < 8; ++t) {
      const _Float16* q = Ht + (size_t)(b * 128 + t * 16 + lr) * NSEQ + k + koffB;
      const v16h bt = *(const v16h*)q;              // 32B contiguous per lane
      acc1[0][t] = wmma_f16(a1[0], bt, acc1[0][t]);
      acc1[1][t] = wmma_f16(a1[1], bt, acc1[1][t]);
    }
    // ---- f2 column tile (batch b, 16 cols) ----
    {
      const _Float16* q = Ht + (size_t)(1024 + b * 16 + lr) * NSEQ + k + koffB;
      const v16h bt = *(const v16h*)q;
      acc2[0] = wmma_f16(a2[0], bt, acc2[0]);
      acc2[1] = wmma_f16(a2[1], bt, acc2[1]);
    }
  }

  // ---- epilogue: per 16-row half, feats->LDS(f16) then out = feats@Wbt+bb ----
  #pragma unroll
  for (int rt = 0; rt < 2; ++rt) {
    __syncthreads();
    // scatter C/D layout into per-batch padded rows: fl[(row*8+b)*160 + kk]
    #pragma unroll
    for (int r = 0; r < 8; ++r) {
      const int row = r + 8 * half;                 // 0..15 within tile
      _Float16* base = &fl[(row * BATCH + b) * KPAD];
      #pragma unroll
      for (int t = 0; t < 8; ++t) base[t * 16 + lr] = (_Float16)acc1[rt][t][r];
      base[128 + lr] = (_Float16)acc2[rt][r];
      base[144 + lr] = (_Float16)0.0f;              // K padding 144..159
    }
    __syncthreads();

    v8f oacc[8];
    #pragma unroll
    for (int t2 = 0; t2 < 8; ++t2) oacc[t2] = (v8f){};

    #pragma unroll
    for (int s = 0; s < KPAD / 32; ++s) {           // K = 160 -> 5 steps
      // A fragment from LDS feats (row = lr, contiguous K runs)
      const _Float16* pa = &fl[(lr * BATCH + b) * KPAD + s * 32 + koffA];
      const h8 alo = *(const h8*)pa;                // K +0..7 of lane group
      const h8 ahi = *(const h8*)(pa + 8);          // K +8..15 of lane group
      v16h av;
      #pragma unroll
      for (int e = 0; e < 8; ++e) { av[e] = alo[e]; av[8 + e] = ahi[e]; }
      #pragma unroll
      for (int t2 = 0; t2 < 8; ++t2) {
        const _Float16* pb = Wbt + (size_t)(t2 * 16 + lr) * KPAD + s * 32 + koffB;
        const v16h bv = *(const v16h*)pb;
        oacc[t2] = wmma_f16(av, bv, oacc[t2]);
      }
    }
    // store out[b][ibase+rt*16+row][n] + bb[n]
    #pragma unroll
    for (int t2 = 0; t2 < 8; ++t2) {
      const float bias = bb[t2 * 16 + lr];
      #pragma unroll
      for (int r = 0; r < 8; ++r) {
        const int row = r + 8 * half;
        const int gi  = ibase + rt * 16 + row;
        out[((size_t)b * NSEQ + gi) * CDIM + t2 * 16 + lr] = oacc[t2][r] + bias;
      }
    }
  }
}

extern "C" void kernel_launch(void* const* d_in, const int* in_sizes, int n_in,
                              void* d_out, int out_size, void* d_ws, size_t ws_size,
                              hipStream_t stream) {
  const float* x  = (const float*)d_in[0];
  const int*   sp = (const int*)  d_in[1];
  const float* W1 = (const float*)d_in[2];
  const float* b1 = (const float*)d_in[3];
  const float* W2 = (const float*)d_in[4];
  const float* b2 = (const float*)d_in[5];
  const float* Wb = (const float*)d_in[6];
  const float* bb = (const float*)d_in[7];
  float* out = (float*)d_out;

  _Float16* Ht  = (_Float16*)d_ws;                               // 1152*4096*2 = 9.44 MB
  _Float16* Wbt = (_Float16*)((char*)d_ws + (size_t)FCOLS * NSEQ * 2); // +40 KB

  prep_ht<<<BATCH * NSEQ, 160, 0, stream>>>(x, W1, b1, W2, b2, Ht);
  prep_wbt<<<(128 * KPAD + 255) / 256, 256, 0, stream>>>(Wb, Wbt);
  masked_gemm_fused<<<NSEQ / 32, 256, 0, stream>>>(sp, Ht, Wbt, bb, out);
}
